// ArcMarginProduct_20916490732065
// MI455X (gfx1250) — matrix operand order repaired and checked by reading, not compiled, not run on hardware
//
#include <hip/hip_runtime.h>
#include <hip/hip_bf16.h>
#include <math.h>

typedef __attribute__((ext_vector_type(16))) _Float16 v16h;
typedef __attribute__((ext_vector_type(8)))  _Float16 v8h;
typedef __attribute__((ext_vector_type(8)))  float    v8f;

#define B_ 1024
#define D_ 512
#define V_ 100000
#define MARGIN 0.4f
#define EPS_ 1e-7f

#define LDS_COL_H 520            // halves per column in LDS (512 + 8 pad -> 1040 B stride, bank-conflict free)
#define TILE_LDS_H (16 * LDS_COL_H)

#if __has_builtin(__builtin_amdgcn_sched_group_barrier)
#define SCHED_GROUP(mask, size) __builtin_amdgcn_sched_group_barrier((mask), (size), 0)
#else
#define SCHED_GROUP(mask, size)
#endif

// ---------- kernel 1: per-column L2 norm of weight (D x V, row-major) ----------
__global__ void colnorm_kernel(const float* __restrict__ w, float* __restrict__ invw) {
    int v = blockIdx.x * blockDim.x + threadIdx.x;
    if (v >= V_) return;
    const float* p = w + v;
    float s = 0.f;
#pragma unroll 8
    for (int d = 0; d < D_; ++d) {
        float t = p[(size_t)d * V_];
        s += t * t;
    }
    invw[v] = 1.f / fmaxf(sqrtf(s), 1e-12f);
}

// ---------- kernel 2: transpose + normalize + f16 convert: wT[v][d] = w[d][v]/||col v|| ----------
__global__ void wtranspose_kernel(const float* __restrict__ w, const float* __restrict__ invw,
                                  _Float16* __restrict__ wT) {
    __shared__ float tile[32][33];
    const int v0 = blockIdx.x * 32;      // V_/32 = 3125 exact
    const int d0 = blockIdx.y * 32;      // D_/32 = 16  exact
    const int tx = threadIdx.x, ty = threadIdx.y;   // 32 x 8
#pragma unroll
    for (int j = 0; j < 4; ++j)
        tile[ty + 8 * j][tx] = w[(size_t)(d0 + ty + 8 * j) * V_ + (v0 + tx)];
    __syncthreads();
#pragma unroll
    for (int j = 0; j < 4; ++j) {
        int v = v0 + ty + 8 * j;
        float inv = invw[v];
        wT[(size_t)v * D_ + (d0 + tx)] = (_Float16)(tile[tx][ty + 8 * j] * inv);
    }
}

// ---------- kernel 3: per-row L2-normalize x to f16, also zero S ----------
__global__ void xnorm_kernel(const float* __restrict__ x, _Float16* __restrict__ xh,
                             float* __restrict__ S) {
    __shared__ float red[256];
    const int b = blockIdx.x;
    const int t = threadIdx.x;
    float a0 = x[(size_t)b * D_ + t];
    float a1 = x[(size_t)b * D_ + t + 256];
    red[t] = a0 * a0 + a1 * a1;
    __syncthreads();
    for (int s = 128; s > 0; s >>= 1) { if (t < s) red[t] += red[t + s]; __syncthreads(); }
    float inv = 1.f / fmaxf(sqrtf(red[0]), 1e-12f);
    xh[(size_t)b * D_ + t]       = (_Float16)(a0 * inv);
    xh[(size_t)b * D_ + t + 256] = (_Float16)(a1 * inv);
    if (t == 0) S[b] = 0.f;
}

// Stage one 16-col x 512-K f16 B-tile (16 KB) from global wT into LDS, split over 4 waves.
// 8 async b128 copies per wave; columns get a 16 B pad (stride 1040 B) for bank-conflict-free reads.
__device__ __forceinline__ void stage_tile(uint64_t wbase, int n0, uint32_t ldsDst,
                                           int wave, int lane) {
#pragma unroll
    for (int i = 0; i < 8; ++i) {
        int chunk = wave * 256 + i * 32 + lane;           // 0..1023, 16 B each
        int col    = chunk >> 6;                          // 0..15
        int within = (chunk & 63) << 4;                   // byte offset inside 1024 B column
        uint32_t goff  = (uint32_t)(n0 + col) * 1024u + (uint32_t)within;
        uint32_t laddr = ldsDst + (uint32_t)col * 1040u + (uint32_t)within;
        asm volatile("global_load_async_to_lds_b128 %0, %1, %2"
                     :: "v"(laddr), "v"(goff), "s"(wbase)
                     : "memory");
    }
}

// Load one 32-halves B fragment (col = ln, k = s*32 + 8*hi .. +7 and s*32 + 16 + 8*hi .. +7) from the padded LDS tile.
__device__ __forceinline__ v16h load_bfrag(const _Float16* bb, int ln, int hi, int s) {
    const _Float16* bp = bb + ln * LDS_COL_H + s * 32 + 8 * hi;
    v8h blo = *(const v8h*)(bp);
    v8h bhi = *(const v8h*)(bp + 16);
    return __builtin_shufflevector(blo, bhi, 0, 1, 2, 3, 4, 5, 6, 7,
                                   8, 9, 10, 11, 12, 13, 14, 15);
}

// ---------- kernel 4: fused WMMA GEMM + exp-sum, async double-buffered B in LDS ----------
// grid = (B_/64, NCHUNK), block = 128 (4 waves). All 4 waves share one B tile (16 cols),
// each wave owns a 16-row A slab held entirely in registers (16 x v16h = 128 VGPRs).
// The k-loop is software-pipelined; sched_group_barrier pins DS reads 2 fragments ahead
// of the consuming WMMA so the waitcnt pass emits s_wait_dscnt > 0 (latency hiding).
__global__ void __launch_bounds__(128)
gemm_exp_kernel(const _Float16* __restrict__ xh, const _Float16* __restrict__ wT,
                float* __restrict__ S) {
    __shared__ __align__(128) _Float16 bbuf[2][TILE_LDS_H];

    const int lane = threadIdx.x & 31;
    const int wave = threadIdx.x >> 5;
    const int hi   = lane >> 4;          // half-wave select
    const int ln   = lane & 15;
    const int rowW = blockIdx.x * 64 + wave * 16;   // this wave's 16-row slab

    // ---- preload A fragments for all K (16 steps of 32) into registers ----
    v16h A[16];
    const _Float16* aBase = xh + (size_t)(rowW + ln) * D_;
#pragma unroll
    for (int s = 0; s < 16; ++s) {
        const int k0 = s * 32;
        v8h alo = *(const v8h*)(aBase + k0 + 8 * hi);
        v8h ahi = *(const v8h*)(aBase + k0 + 8 * hi + 16);
        A[s] = __builtin_shufflevector(alo, ahi, 0, 1, 2, 3, 4, 5, 6, 7,
                                       8, 9, 10, 11, 12, 13, 14, 15);
    }

    float sum[8];
#pragma unroll
    for (int r = 0; r < 8; ++r) sum[r] = 0.f;

    const uint64_t wbase = (uint64_t)(uintptr_t)wT;
    const uint32_t lds0  = (uint32_t)(uintptr_t)&bbuf[0][0];
    const uint32_t bufBytes = TILE_LDS_H * 2;

    const int nTiles = V_ / 16;          // 6250 exact
    const int step   = gridDim.y;
    int t = blockIdx.y;
    int bufIdx = 0;

    // prologue: stage first tile into buffer 0
    stage_tile(wbase, t * 16, lds0, wave, lane);

    for (;;) {
        const int tn = t + step;
        const bool hasNext = tn < nTiles;
        if (hasNext) {
            stage_tile(wbase, tn * 16, lds0 + (uint32_t)(bufIdx ^ 1) * bufBytes, wave, lane);
            asm volatile("s_wait_asynccnt 0x8" ::: "memory");   // retire current tile's 8 copies
        } else {
            asm volatile("s_wait_asynccnt 0x0" ::: "memory");
        }
        __syncthreads();                                        // tile visible to all waves

        const _Float16* bb = &bbuf[bufIdx][0];
        v8f acc = {};
        // software-pipelined k-loop: fragment loads run 2 steps ahead of the WMMA consuming them
        v16h bfrag[3];
        bfrag[0] = load_bfrag(bb, ln, hi, 0);
        bfrag[1] = load_bfrag(bb, ln, hi, 1);
#pragma unroll
        for (int s = 0; s < 16; ++s) {
            if (s + 2 < 16) bfrag[(s + 2) % 3] = load_bfrag(bb, ln, hi, s + 2);
            acc = __builtin_amdgcn_wmma_f32_16x16x32_f16(
                false, A[s], false, bfrag[s % 3], (short)0, acc, false, false);
        }
        // pin the pipeline: 4 ds_read up front, then {1 WMMA, 2 ds_read} x14, then 2 WMMA
        SCHED_GROUP(0x100, 4);            // DS read group (frags 0,1)
#pragma unroll
        for (int s = 0; s < 14; ++s) {
            SCHED_GROUP(0x008, 1);        // WMMA (MFMA class) consuming frag s
            SCHED_GROUP(0x100, 2);        // DS reads prefetching frag s+2
        }
        SCHED_GROUP(0x008, 1);
        SCHED_GROUP(0x008, 1);

        // C layout: VGPR r -> row (rowW + 8*hi + r), col (t*16 + ln)
#pragma unroll
        for (int r = 0; r < 8; ++r) {
            float c = fminf(fmaxf(acc[r], -1.f + EPS_), 1.f - EPS_);
            sum[r] += __expf(c);
        }
        __syncthreads();                                        // safe to overwrite this buffer
        if (!hasNext) break;
        t = tn;
        bufIdx ^= 1;
    }

    // reduce each row accumulator across the 16 lanes of its half-wave
#pragma unroll
    for (int r = 0; r < 8; ++r) {
        float v = sum[r];
        v += __shfl_xor(v, 1, 32);
        v += __shfl_xor(v, 2, 32);
        v += __shfl_xor(v, 4, 32);
        v += __shfl_xor(v, 8, 32);
        if (ln == 0) atomicAdd(&S[rowW + 8 * hi + r], v);
    }
}

// ---------- kernel 5: label-column cosines (one wave per row) ----------
__global__ void clab_kernel(const _Float16* __restrict__ xh, const _Float16* __restrict__ wT,
                            const int* __restrict__ labels, float* __restrict__ clab) {
    const int wid  = (blockIdx.x * blockDim.x + threadIdx.x) >> 5;
    const int lane = threadIdx.x & 31;
    if (wid >= B_) return;
    const int lab = labels[wid];
    const _Float16* xp = xh + (size_t)wid * D_;
    const _Float16* wp = wT + (size_t)lab * D_;
    float s = 0.f;
#pragma unroll 4
    for (int d = lane; d < D_; d += 32) s += (float)xp[d] * (float)wp[d];
#pragma unroll
    for (int off = 16; off; off >>= 1) s += __shfl_xor(s, off, 32);
    if (lane == 0) clab[wid] = s;
}

// ---------- kernel 6: margin + logsumexp correction + mean ----------
__global__ void loss_kernel(const float* __restrict__ S, const float* __restrict__ clab,
                            float* __restrict__ out) {
    __shared__ float red[256];
    const int t = threadIdx.x;
    float acc = 0.f;
    for (int b = t; b < B_; b += 256) {
        float c  = fminf(fmaxf(clab[b], -1.f + EPS_), 1.f - EPS_);
        float tt = cosf(acosf(c) + MARGIN);
        float Z  = S[b] - __expf(c) + __expf(tt);
        acc += logf(Z) - tt;
    }
    red[t] = acc;
    __syncthreads();
    for (int s = 128; s > 0; s >>= 1) { if (t < s) red[t] += red[t + s]; __syncthreads(); }
    if (t == 0) out[0] = red[0] / (float)B_;
}

extern "C" void kernel_launch(void* const* d_in, const int* in_sizes, int n_in,
                              void* d_out, int out_size, void* d_ws, size_t ws_size,
                              hipStream_t stream) {
    (void)in_sizes; (void)n_in; (void)out_size; (void)ws_size;
    const float* x      = (const float*)d_in[0];   // [B, D] f32
    const float* w      = (const float*)d_in[1];   // [D, V] f32
    const int*   labels = (const int*)d_in[2];     // [B]

    char* ws = (char*)d_ws;
    size_t off = 0;
    _Float16* wT   = (_Float16*)(ws + off); off += (size_t)V_ * D_ * sizeof(_Float16);
    off = (off + 255) & ~(size_t)255;
    float*    invw = (float*)(ws + off);    off += (size_t)V_ * sizeof(float);
    off = (off + 255) & ~(size_t)255;
    _Float16* xh   = (_Float16*)(ws + off); off += (size_t)B_ * D_ * sizeof(_Float16);
    off = (off + 255) & ~(size_t)255;
    float*    Ssum = (float*)(ws + off);    off += (size_t)B_ * sizeof(float);
    off = (off + 255) & ~(size_t)255;
    float*    clab = (float*)(ws + off);    off += (size_t)B_ * sizeof(float);

    colnorm_kernel<<<(V_ + 255) / 256, 256, 0, stream>>>(w, invw);
    wtranspose_kernel<<<dim3(V_ / 32, D_ / 32), dim3(32, 8), 0, stream>>>(w, invw, wT);
    xnorm_kernel<<<B_, 256, 0, stream>>>(x, xh, Ssum);
    gemm_exp_kernel<<<dim3(B_ / 64, 128), 128, 0, stream>>>(xh, wT, Ssum);
    clab_kernel<<<(B_ * 32) / 256, 256, 0, stream>>>(xh, wT, labels, clab);
    loss_kernel<<<1, 256, 0, stream>>>(Ssum, clab, (float*)d_out);
}
